// WinDecoderTransformer_70514773066141
// MI455X (gfx1250) — matrix operand order repaired and checked by reading, compiled, not run on hardware
//
#include <hip/hip_runtime.h>
#include <hip/hip_bf16.h>

// ---------------------------------------------------------------------------
// WinDecoderTransformer for MI455X (gfx1250, wave32, WMMA bf16)
// Stage: window gather -> [2x decoder layer] -> final LN -> window scatter.
// All GEMM-shaped math via v_wmma_f32_16x16x32_bf16 (fp32 accumulate),
// double-buffered LDS staging with vectorized b128 global loads.
// ---------------------------------------------------------------------------

typedef __attribute__((ext_vector_type(16))) __bf16 v16bf;
typedef __attribute__((ext_vector_type(8)))  __bf16 v8bf;
typedef __attribute__((ext_vector_type(8)))  float  v8f;

union Frag { v16bf v; v8bf h[2]; };

#define C_DIM   256
#define NWIN    512
#define LQ      128
#define LKM     32
#define MQ      (NWIN * LQ)     // 65536 tgt tokens
#define MMEM    (NWIN * LKM)    // 16384 mem tokens
#define NEGV    (-1000000000.0f)

// ---------------------------------------------------------------------------
// Shifted-window region ids (stage 1 only); mirror get_attn_mask() slices.
// ---------------------------------------------------------------------------
__device__ __forceinline__ int region_tgt_s1(int n, int l) {
    int hi = n >> 4, wi = n & 15;       // nH=32, nW=16
    int i  = l >> 4, j  = l & 15;       // win 8x16
    int h  = hi * 8 + i, w = wi * 16 + j;
    int rh = (h < 248) ? 0 : ((h < 252) ? 1 : 2);
    int rw = (w < 240) ? 0 : ((w < 248) ? 1 : 2);
    return rh * 3 + rw;
}
__device__ __forceinline__ int region_mem_s1(int n, int l) {
    int hi = n >> 4, wi = n & 15;       // nH=32, nW=16
    int i  = l >> 3, j  = l & 7;        // win 4x8
    int h  = hi * 4 + i, w = wi * 8 + j;
    int rh = (h < 124) ? 0 : ((h < 126) ? 1 : 2);
    int rw = (w < 120) ? 0 : ((w < 124) ? 1 : 2);
    return rh * 3 + rw;
}

// ---------------------------------------------------------------------------
// GEMM:  D[M,N] = (A[M,K] (+ Aadd)) @ W[N,K]^T + bias  (optional ReLU)
// 128x128 tile / 256-thread block (8 waves, 32x64 each), bf16 WMMA,
// double-buffered LDS, b128 global loads, b128 LDS stores.
// ---------------------------------------------------------------------------
template<bool HASADD, bool RELU>
__global__ __launch_bounds__(256)
void gemm_kernel(const float* __restrict__ A, const float* __restrict__ Aadd,
                 const float* __restrict__ W, const float* __restrict__ bias,
                 float* __restrict__ D, int M, int N, int K)
{
    __shared__ __align__(16) __bf16 As[2][128][40];
    __shared__ __align__(16) __bf16 Bs[2][128][40];

    const int tid  = threadIdx.x;
    const int lane = tid & 31;
    const int wave = tid >> 5;
    const int lr   = lane & 15;
    const int half = lane >> 4;
    const int tileM = blockIdx.x * 128;
    const int tileN = blockIdx.y * 128;
    const int wm = (wave & 3) * 32;
    const int wn = (wave >> 2) * 64;
    const int ko = half ? 8  : 0;   // A-frag K chunk base
    const int kb = half ? 8  : 0;   // B-frag K base

    // Staging: thread -> (row = tid/2, 16-col half = (tid&1)*16)
    const int srow = tid >> 1;
    const int scol = (tid & 1) * 16;
    const float* Arow = A + (size_t)(tileM + srow) * K + scol;
    const float* Prow = HASADD ? (Aadd + (size_t)(tileM + srow) * K + scol) : nullptr;
    const float* Wrow = W + (size_t)(tileN + srow) * K + scol;

    float fa[16], fb[16];
    auto loadT = [&](int k0) {
#pragma unroll
        for (int q = 0; q < 4; ++q) {
            *(float4*)&fa[q * 4] = *(const float4*)(Arow + k0 + q * 4);
            *(float4*)&fb[q * 4] = *(const float4*)(Wrow + k0 + q * 4);
        }
        if (HASADD) {
#pragma unroll
            for (int q = 0; q < 4; ++q) {
                float4 p = *(const float4*)(Prow + k0 + q * 4);
                fa[q*4+0] += p.x; fa[q*4+1] += p.y;
                fa[q*4+2] += p.z; fa[q*4+3] += p.w;
            }
        }
    };
    auto storeT = [&](int buf) {
        __bf16 ta[16], tb[16];
#pragma unroll
        for (int i = 0; i < 16; ++i) { ta[i] = (__bf16)fa[i]; tb[i] = (__bf16)fb[i]; }
        *(v8bf*)&As[buf][srow][scol]     = *(v8bf*)&ta[0];
        *(v8bf*)&As[buf][srow][scol + 8] = *(v8bf*)&ta[8];
        *(v8bf*)&Bs[buf][srow][scol]     = *(v8bf*)&tb[0];
        *(v8bf*)&Bs[buf][srow][scol + 8] = *(v8bf*)&tb[8];
    };

    v8f acc[2][4];
#pragma unroll
    for (int i = 0; i < 2; ++i)
#pragma unroll
        for (int j = 0; j < 4; ++j) acc[i][j] = {};

    const int NK = K >> 5;
    loadT(0);
    storeT(0);
    __syncthreads();

    for (int kt = 0; kt < NK; ++kt) {
        const int cur = kt & 1;
        if (kt + 1 < NK) loadT((kt + 1) << 5);   // overlap with WMMA burst

        Frag a[2], b[4];
#pragma unroll
        for (int i = 0; i < 2; ++i) {
            int r = wm + i * 16 + lr;
            a[i].h[0] = *(const v8bf*)&As[cur][r][ko];
            a[i].h[1] = *(const v8bf*)&As[cur][r][ko + 16];
        }
#pragma unroll
        for (int j = 0; j < 4; ++j) {
            int c = wn + j * 16 + lr;
            b[j].h[0] = *(const v8bf*)&Bs[cur][c][kb];
            b[j].h[1] = *(const v8bf*)&Bs[cur][c][kb + 16];
        }
#pragma unroll
        for (int i = 0; i < 2; ++i)
#pragma unroll
            for (int j = 0; j < 4; ++j)
                acc[i][j] = __builtin_amdgcn_wmma_f32_16x16x32_bf16(
                    false, a[i].v, false, b[j].v, (short)0, acc[i][j], false, false);

        if (kt + 1 < NK) storeT(cur ^ 1);
        __syncthreads();
    }

    // Epilogue: bias / relu / store
#pragma unroll
    for (int i = 0; i < 2; ++i) {
#pragma unroll
        for (int j = 0; j < 4; ++j) {
            int gn = tileN + wn + j * 16 + lr;
            float bv = bias ? bias[gn] : 0.0f;
#pragma unroll
            for (int e = 0; e < 8; ++e) {
                int gm = tileM + wm + i * 16 + e + half * 8;
                float v = acc[i][j][e] + bv;
                if (RELU) v = v > 0.0f ? v : 0.0f;
                D[(size_t)gm * N + gn] = v;
            }
        }
    }
}

// ---------------------------------------------------------------------------
// Fused attention per (window, head): scores WMMA -> analytic shifted-window
// mask -> softmax -> P@V WMMA.  Output overwrites Q in place.
// maskmode: 0=none, 1=self(stage1), 2=cross(stage1)
// ---------------------------------------------------------------------------
template<int LK>
__global__ __launch_bounds__(256)
void attn_kernel(float* __restrict__ Q, const float* __restrict__ Kb,
                 const float* __restrict__ Vb, int maskmode)
{
    constexpr int NT = LK / 16;
    constexpr int PS = LK + 8;
    __shared__ __align__(16) __bf16 Qs[128][40];
    __shared__ __align__(16) __bf16 Ks[LK][40];
    __shared__ __align__(16) __bf16 Vt[32][PS];
    __shared__ __align__(16) __bf16 Ps[128][PS];

    const int tid  = threadIdx.x;
    const int lane = tid & 31;
    const int wave = tid >> 5;
    const int lr   = lane & 15;
    const int half = lane >> 4;
    const int n = blockIdx.x >> 3;
    const int h = blockIdx.x & 7;
    const size_t qbase = (size_t)n * LQ * C_DIM + h * 32;
    const size_t kbase = (size_t)n * LK * C_DIM + h * 32;

    // Stage Q (vectorized: one 16-float span per thread)
    {
        int row = tid >> 1, c0 = (tid & 1) * 16;
        float fq[16];
#pragma unroll
        for (int q = 0; q < 4; ++q)
            *(float4*)&fq[q * 4] = *(const float4*)(Q + qbase + (size_t)row * C_DIM + c0 + q * 4);
        __bf16 t[16];
#pragma unroll
        for (int i = 0; i < 16; ++i) t[i] = (__bf16)fq[i];
        *(v8bf*)&Qs[row][c0]     = *(v8bf*)&t[0];
        *(v8bf*)&Qs[row][c0 + 8] = *(v8bf*)&t[8];
    }
    // Stage K (row-major) and V (transposed scatter)
    for (int it = tid; it < LK * 2; it += 256) {
        int row = it >> 1, c0 = (it & 1) * 16;
        float fk[16], fv[16];
#pragma unroll
        for (int q = 0; q < 4; ++q) {
            *(float4*)&fk[q * 4] = *(const float4*)(Kb + kbase + (size_t)row * C_DIM + c0 + q * 4);
            *(float4*)&fv[q * 4] = *(const float4*)(Vb + kbase + (size_t)row * C_DIM + c0 + q * 4);
        }
        __bf16 t[16];
#pragma unroll
        for (int i = 0; i < 16; ++i) t[i] = (__bf16)fk[i];
        *(v8bf*)&Ks[row][c0]     = *(v8bf*)&t[0];
        *(v8bf*)&Ks[row][c0 + 8] = *(v8bf*)&t[8];
#pragma unroll
        for (int i = 0; i < 16; ++i) Vt[c0 + i][row] = (__bf16)fv[i];
    }
    __syncthreads();

    const int r0 = wave * 16;
    const int ko = half ? 8 : 0, kb2 = half ? 8 : 0;

    Frag aq;
    aq.h[0] = *(const v8bf*)&Qs[r0 + lr][ko];
    aq.h[1] = *(const v8bf*)&Qs[r0 + lr][ko + 16];

    v8f S[NT];
#pragma unroll
    for (int t = 0; t < NT; ++t) S[t] = {};
#pragma unroll
    for (int t = 0; t < NT; ++t) {
        Frag bk;
        bk.h[0] = *(const v8bf*)&Ks[t * 16 + lr][kb2];
        bk.h[1] = *(const v8bf*)&Ks[t * 16 + lr][kb2 + 16];
        S[t] = __builtin_amdgcn_wmma_f32_16x16x32_bf16(
            false, aq.v, false, bk.v, (short)0, S[t], false, false);
    }
    const float scale = 0.17677669529663687f;   // 1/sqrt(32)
#pragma unroll
    for (int t = 0; t < NT; ++t)
#pragma unroll
        for (int e = 0; e < 8; ++e) S[t][e] *= scale;

    if (maskmode) {
        int qreg[8];
#pragma unroll
        for (int e = 0; e < 8; ++e) qreg[e] = region_tgt_s1(n, r0 + e + half * 8);
#pragma unroll
        for (int t = 0; t < NT; ++t) {
            int lk = t * 16 + lr;
            int kreg = (maskmode == 1) ? region_tgt_s1(n, lk) : region_mem_s1(n, lk);
#pragma unroll
            for (int e = 0; e < 8; ++e)
                if (qreg[e] != kreg) S[t][e] += NEGV;
        }
    }

    // Row-wise softmax (row = e + 8*half within the wave's 16-row slab)
#pragma unroll
    for (int e = 0; e < 8; ++e) {
        float m = S[0][e];
#pragma unroll
        for (int t = 1; t < NT; ++t) m = fmaxf(m, S[t][e]);
        for (int o = 1; o < 16; o <<= 1) m = fmaxf(m, __shfl_xor(m, o, 32));
        float s = 0.0f;
#pragma unroll
        for (int t = 0; t < NT; ++t) { float p = __expf(S[t][e] - m); S[t][e] = p; s += p; }
        for (int o = 1; o < 16; o <<= 1) s += __shfl_xor(s, o, 32);
        float inv = 1.0f / s;
        int row = r0 + e + half * 8;
#pragma unroll
        for (int t = 0; t < NT; ++t)
            Ps[row][t * 16 + lr] = (__bf16)(S[t][e] * inv);
    }
    __syncthreads();

    v8f O[2]; O[0] = {}; O[1] = {};
#pragma unroll
    for (int ks = 0; ks < LK / 32; ++ks) {
        Frag ap, bv0, bv1;
        ap.h[0]  = *(const v8bf*)&Ps[r0 + lr][ks * 32 + ko];
        ap.h[1]  = *(const v8bf*)&Ps[r0 + lr][ks * 32 + ko + 16];
        bv0.h[0] = *(const v8bf*)&Vt[lr][ks * 32 + kb2];
        bv0.h[1] = *(const v8bf*)&Vt[lr][ks * 32 + kb2 + 16];
        bv1.h[0] = *(const v8bf*)&Vt[16 + lr][ks * 32 + kb2];
        bv1.h[1] = *(const v8bf*)&Vt[16 + lr][ks * 32 + kb2 + 16];
        O[0] = __builtin_amdgcn_wmma_f32_16x16x32_bf16(
            false, ap.v, false, bv0.v, (short)0, O[0], false, false);
        O[1] = __builtin_amdgcn_wmma_f32_16x16x32_bf16(
            false, ap.v, false, bv1.v, (short)0, O[1], false, false);
    }
#pragma unroll
    for (int j = 0; j < 2; ++j)
#pragma unroll
        for (int e = 0; e < 8; ++e) {
            int gm = r0 + e + half * 8;
            Q[qbase + (size_t)gm * C_DIM + j * 16 + lr] = O[j][e];
        }
}

// ---------------------------------------------------------------------------
// LayerNorm(+residual): one wave per 256-wide row, float4 traffic.
// ---------------------------------------------------------------------------
template<bool HASRES>
__global__ __launch_bounds__(256)
void ln_kernel(const float* __restrict__ X, const float* __restrict__ R,
               const float* __restrict__ g, const float* __restrict__ b,
               float* __restrict__ Out, int M)
{
    int row  = blockIdx.x * 8 + (threadIdx.x >> 5);
    int lane = threadIdx.x & 31;
    if (row >= M) return;
    size_t base = (size_t)row * C_DIM + lane * 8;
    float v[8];
    *(float4*)&v[0] = *(const float4*)(X + base);
    *(float4*)&v[4] = *(const float4*)(X + base + 4);
    if (HASRES) {
        float4 r0 = *(const float4*)(R + base);
        float4 r1 = *(const float4*)(R + base + 4);
        v[0] += r0.x; v[1] += r0.y; v[2] += r0.z; v[3] += r0.w;
        v[4] += r1.x; v[5] += r1.y; v[6] += r1.z; v[7] += r1.w;
    }
    float s = 0.0f;
#pragma unroll
    for (int i = 0; i < 8; ++i) s += v[i];
    for (int o = 1; o < 32; o <<= 1) s += __shfl_xor(s, o, 32);
    float mu = s * (1.0f / 256.0f);
    float q = 0.0f;
#pragma unroll
    for (int i = 0; i < 8; ++i) { float d = v[i] - mu; q += d * d; }
    for (int o = 1; o < 32; o <<= 1) q += __shfl_xor(q, o, 32);
    float rs = rsqrtf(q * (1.0f / 256.0f) + 1e-5f);
#pragma unroll
    for (int i = 0; i < 8; ++i) {
        int c = lane * 8 + i;
        Out[(size_t)row * C_DIM + c] = (v[i] - mu) * rs * g[c] + b[c];
    }
}

// ---------------------------------------------------------------------------
// Dilated / shifted window (un)partition.  img is (C,H,W); win is [n][l][c].
// ---------------------------------------------------------------------------
struct Geom { int s, H, W, winh, winw, nH, nW, L, rollh, rollw; };

__device__ __forceinline__ void win_coords(const Geom& g, int n, int l,
                                           int& h, int& w)
{
    int nHW = g.nH * g.nW;
    int bd = n / nHW, r2 = n % nHW;
    int hi = r2 / g.nW, wi = r2 % g.nW;
    int shi = bd / g.s, swi = bd % g.s;
    int i = l / g.winw, j = l % g.winw;
    h = (hi * g.winh + i) * g.s + shi;
    w = (wi * g.winw + j) * g.s + swi;
}

__global__ void gather_kernel(const float* __restrict__ img,
                              float* __restrict__ dst, Geom g, int total)
{
    int idx = blockIdx.x * 256 + threadIdx.x;
    if (idx >= total) return;
    int c = idx & 255;
    int rem = idx >> 8;
    int l = rem % g.L, n = rem / g.L;
    int h, w; win_coords(g, n, l, h, w);
    h = (h + g.rollh) % g.H;
    w = (w + g.rollw) % g.W;
    dst[idx] = img[(size_t)c * g.H * g.W + (size_t)h * g.W + w];
}

__global__ void scatter_img_kernel(const float* __restrict__ srcw,
                                   float* __restrict__ img, Geom g, int total)
{
    int idx = blockIdx.x * 256 + threadIdx.x;
    if (idx >= total) return;
    int c = idx & 255;
    int rem = idx >> 8;
    int l = rem % g.L, n = rem / g.L;
    int h, w; win_coords(g, n, l, h, w);
    img[(size_t)c * g.H * g.W + (size_t)h * g.W + w] = srcw[idx];
}

__global__ void scatter_out_kernel(const float* __restrict__ srcw,
                                   float* __restrict__ out, Geom g, int total)
{
    int idx = blockIdx.x * 256 + threadIdx.x;
    if (idx >= total) return;
    int c = idx & 255;
    int rem = idx >> 8;
    int l = rem % g.L, n = rem / g.L;
    int h, w; win_coords(g, n, l, h, w);
    out[(size_t)(h * g.W + w) * C_DIM + c] = srcw[idx];
}

// ---------------------------------------------------------------------------
// Host orchestration
// ---------------------------------------------------------------------------
static inline void gemm(hipStream_t st, const float* A, const float* Aadd,
                        const float* W, const float* bias, float* D,
                        int M, int N, int K, bool relu)
{
    dim3 grid(M / 128, N / 128);
    if (Aadd)
        gemm_kernel<true, false><<<grid, 256, 0, st>>>(A, Aadd, W, bias, D, M, N, K);
    else if (relu)
        gemm_kernel<false, true><<<grid, 256, 0, st>>>(A, nullptr, W, bias, D, M, N, K);
    else
        gemm_kernel<false, false><<<grid, 256, 0, st>>>(A, nullptr, W, bias, D, M, N, K);
}

extern "C" void kernel_launch(void* const* d_in, const int* in_sizes, int n_in,
                              void* d_out, int out_size, void* d_ws, size_t ws_size,
                              hipStream_t stream)
{
    const float* q_feats = (const float*)d_in[0];
    const float* q_embed = (const float*)d_in[1];
    const float* src_in  = (const float*)d_in[2];
    const float* pos_emb = (const float*)d_in[3];
    // d_in[4]=mask (all false), d_in[5]=div : unused
    const float* points  = (const float*)d_in[6];

    // Workspace arena (floats)
    float* ws   = (float*)d_ws;
    float* TGT  = ws;                       // 16,777,216
    float* QPOS = ws + 16777216;            // 16,777,216
    float* MEMB = ws + 33554432;            //  4,194,304
    float* MPOS = ws + 37748736;            //  4,194,304
    float* B1   = ws + 41943040;            // 16,777,216
    float* B2   = ws + 58720256;            // 16,777,216  (B2+B3 = FFN hidden)
    float* B3   = ws + 75497472;            // 16,777,216
    (void)B3; (void)ws_size; (void)n_in;

    auto pp = [&](int i) { return (const float*)d_in[i]; };
    // params pytree (sorted-key) leaves at d_in[7..44]:
    // per layer: ca.{in_b,in_w,out_b,out_w}, l1b,l1w,l2b,l2w,
    //            n1b,n1w,n2b,n2w,n3b,n3w, sa.{in_b,in_w,out_b,out_w}
    const int PBASE = 7;
    const float* norm_b = pp(PBASE + 36);
    const float* norm_w = pp(PBASE + 37);

    const Geom GT0 = {2, 256, 256, 8, 16, 16,  8, 128, 0, 0};
    const Geom GM0 = {2, 128, 128, 4,  8, 16,  8,  32, 0, 0};
    const Geom GT1 = {1, 256, 256, 8, 16, 32, 16, 128, 4, 8};
    const Geom GM1 = {1, 128, 128, 4,  8, 32, 16,  32, 2, 4};

    for (int st = 0; st < 2; ++st) {
        Geom gt = st ? GT1 : GT0;
        Geom gm = st ? GM1 : GM0;
        const float* timg = st ? (const float*)B1 : q_feats;

        gather_kernel<<<MQ * C_DIM / 256, 256, 0, stream>>>(timg,    TGT,  gt, MQ * C_DIM);
        gather_kernel<<<MQ * C_DIM / 256, 256, 0, stream>>>(q_embed, QPOS, gt, MQ * C_DIM);
        gather_kernel<<<MMEM * C_DIM / 256, 256, 0, stream>>>(src_in,  MEMB, gm, MMEM * C_DIM);
        gather_kernel<<<MMEM * C_DIM / 256, 256, 0, stream>>>(pos_emb, MPOS, gm, MMEM * C_DIM);

        for (int ly = 0; ly < 2; ++ly) {
            int pb = PBASE + 18 * ly;
            const float *ca_in_b = pp(pb+0),  *ca_in_w = pp(pb+1);
            const float *ca_out_b = pp(pb+2), *ca_out_w = pp(pb+3);
            const float *l1b = pp(pb+4), *l1w = pp(pb+5);
            const float *l2b = pp(pb+6), *l2w = pp(pb+7);
            const float *n1b = pp(pb+8),  *n1w = pp(pb+9);
            const float *n2b = pp(pb+10), *n2w = pp(pb+11);
            const float *n3b = pp(pb+12), *n3w = pp(pb+13);
            const float *sa_in_b = pp(pb+14), *sa_in_w = pp(pb+15);
            const float *sa_out_b = pp(pb+16), *sa_out_w = pp(pb+17);

            // ---- self attention (q_in = k_in = tgt+qpos, v_in = tgt) ----
            gemm(stream, TGT, QPOS,    sa_in_w,          sa_in_b,       B1, MQ, 256, 256, false);
            gemm(stream, TGT, QPOS,    sa_in_w + 65536,  sa_in_b + 256, B2, MQ, 256, 256, false);
            gemm(stream, TGT, nullptr, sa_in_w + 131072, sa_in_b + 512, B3, MQ, 256, 256, false);
            attn_kernel<128><<<NWIN * 8, 256, 0, stream>>>(B1, B2, B3, st ? 1 : 0);
            gemm(stream, B1, nullptr, sa_out_w, sa_out_b, B2, MQ, 256, 256, false);
            ln_kernel<true><<<MQ / 8, 256, 0, stream>>>(TGT, B2, n1w, n1b, TGT, MQ);

            // ---- cross attention (k_in = mem+pos, v_in = mem) ----
            gemm(stream, TGT,  QPOS,    ca_in_w,          ca_in_b,       B1, MQ,   256, 256, false);
            gemm(stream, MEMB, MPOS,    ca_in_w + 65536,  ca_in_b + 256, B2, MMEM, 256, 256, false);
            gemm(stream, MEMB, nullptr, ca_in_w + 131072, ca_in_b + 512, B3, MMEM, 256, 256, false);
            attn_kernel<32><<<NWIN * 8, 256, 0, stream>>>(B1, B2, B3, st ? 2 : 0);
            gemm(stream, B1, nullptr, ca_out_w, ca_out_b, B2, MQ, 256, 256, false);
            ln_kernel<true><<<MQ / 8, 256, 0, stream>>>(TGT, B2, n2w, n2b, TGT, MQ);

            // ---- FFN (hidden lives in B2..B3 as one 512-wide buffer) ----
            gemm(stream, TGT, nullptr, l1w, l1b, B2, MQ, 512, 256, true);
            gemm(stream, B2,  nullptr, l2w, l2b, B1, MQ, 256, 512, false);
            ln_kernel<true><<<MQ / 8, 256, 0, stream>>>(TGT, B1, n3w, n3b, TGT, MQ);
        }

        // final decoder LayerNorm (per stage, no residual)
        ln_kernel<false><<<MQ / 8, 256, 0, stream>>>(TGT, nullptr, norm_w, norm_b, TGT, MQ);

        if (st == 0) {
            Geom gs = GT0;                  // roll already 0
            scatter_img_kernel<<<MQ * C_DIM / 256, 256, 0, stream>>>(TGT, B1, gs, MQ * C_DIM);
        } else {
            Geom gs = GT1; gs.rollh = 0; gs.rollw = 0;
            scatter_out_kernel<<<MQ * C_DIM / 256, 256, 0, stream>>>(TGT, (float*)d_out, gs, MQ * C_DIM);
        }
    }

    // points_queries pass-through, appended after hs
    int npts = in_sizes[6];
    hipMemcpyAsync((float*)d_out + (out_size - npts), points,
                   (size_t)npts * sizeof(float), hipMemcpyDeviceToDevice, stream);
}
